// Model_51075751084442
// MI455X (gfx1250) — matrix operand order repaired and checked by reading, not compiled, not run on hardware
//
#include <hip/hip_runtime.h>
#include <math.h>

// ---- problem dims ----
#define DD      64
#define EE      4
#define HIDD    256
#define NCLSS   10
#define BATCHH  128
#define SEQQ    1024
#define INN     192
#define NTOKK   (BATCHH * SEQQ)   // 131072
#define NTILEE  (NTOKK / 16)      // 8192 tiles of 16 tokens

// ---- workspace byte offsets (16B-aligned) ----
#define OFF_Z     0         // float zacc[128*64]            (32768 B)
#define OFF_ACC   32768     // float pfacc[8]: p[0..3], f[4..7] (256 B reserved)
#define OFF_PW1T  33024     // bf16 pe_w1^T [64][192]        (24576 B)
#define OFF_PW2T  57600     // bf16 pe_w2^T [64][64]         (8192 B)  -- contiguous after pw1t!
#define OFF_EW1T  65792     // bf16 ew1^T   [4][256][64]     (131072 B)
#define OFF_EW2T  196864    // bf16 ew2^T   [4][64][256]     (131072 B)
// total: 327936 B

#if __has_builtin(__builtin_amdgcn_global_load_async_to_lds_b128)
#define ASYNC_WLOAD 1
#endif

typedef float v8f __attribute__((ext_vector_type(8)));
typedef __bf16 bf16x16 __attribute__((ext_vector_type(16)));
typedef __bf16 bf16x2 __attribute__((ext_vector_type(2)));

// builtin parameter type per hipcc diagnostic: 'int __vector(4) __device__ *'
typedef int v4i_ __attribute__((vector_size(16)));
typedef __attribute__((address_space(1))) void* as1_void;
typedef __attribute__((address_space(3))) void* as3_void;
typedef __attribute__((address_space(1))) v4i_* as1_v4i;
typedef __attribute__((address_space(3))) v4i_* as3_v4i;

union Frag {
    unsigned int u[8];
    bf16x16 v;
};

// scalar RNE convert (prep kernel only — runs once over ~150K elements)
__device__ __forceinline__ unsigned short f2bf(float f) {
    unsigned int u = __builtin_bit_cast(unsigned int, f);
    u += 0x7FFFu + ((u >> 16) & 1u);
    return (unsigned short)(u >> 16);
}

// fast packed convert: two f32 -> packed 2xbf16 in one dword
__device__ __forceinline__ unsigned int pack2(float a, float b) {
#if __has_builtin(__builtin_amdgcn_cvt_pk_bf16_f32)
    bf16x2 p = __builtin_amdgcn_cvt_pk_bf16_f32(a, b);
    return __builtin_bit_cast(unsigned int, p);
#else
    // single v_perm_b32: D = { hi16(b), hi16(a) }  (truncation)
    return __builtin_amdgcn_perm(__builtin_bit_cast(unsigned int, b),
                                 __builtin_bit_cast(unsigned int, a),
                                 0x07060302u);
#endif
}

__device__ __forceinline__ float gelu_exact(float v) {
    return 0.5f * v * (1.0f + erff(v * 0.70710678118654752f));
}
// wave-private LDS write->read fence (cross-lane, same wave)
__device__ __forceinline__ void lds_wave_fence() {
    asm volatile("s_wait_dscnt 0" ::: "memory");
}

__device__ __forceinline__ v8f wmma_bf16(const Frag& a, const Frag& b, v8f c) {
    return __builtin_amdgcn_wmma_f32_16x16x32_bf16(false, a.v, false, b.v,
                                                   (short)0, c, false, false);
}

// B fragment (32x16 bf16, K x N) from bf16 W^T stored [N][K] row-major.
// Works for both global and LDS source pointers (provenance decides vmem vs ds).
__device__ __forceinline__ Frag loadB(const unsigned short* WT,
                                      int ldk, int n0, int k0, int lane) {
    int col = lane & 15, h = lane >> 4;
    const unsigned short* p = WT + (size_t)(n0 + col) * ldk + k0 + 8 * h;
    uint4 a = *(const uint4*)p;
    uint4 b = *(const uint4*)(p + 16);
    Frag f;
    f.u[0] = a.x; f.u[1] = a.y; f.u[2] = a.z; f.u[3] = a.w;
    f.u[4] = b.x; f.u[5] = b.y; f.u[6] = b.z; f.u[7] = b.w;
    return f;
}

// A fragment (16x32 bf16, M x K) from LDS bf16 tile stored [16][ldk] row-major.
__device__ __forceinline__ Frag loadA_lds(const unsigned short* T,
                                          int ldk, int k0, int lane) {
    int row = lane & 15, h = lane >> 4;
    const unsigned short* p = T + row * ldk + k0 + 8 * h;
    uint4 a = *(const uint4*)p;
    uint4 b = *(const uint4*)(p + 16);
    Frag f;
    f.u[0] = a.x; f.u[1] = a.y; f.u[2] = a.z; f.u[3] = a.w;
    f.u[4] = b.x; f.u[5] = b.y; f.u[6] = b.z; f.u[7] = b.w;
    return f;
}

// A fragment from global fp32 rows (convert to bf16 with packed converts)
__device__ __forceinline__ Frag loadA_x(const float* __restrict__ x0,
                                        int ldk, int k0, int lane) {
    int row = lane & 15, h = lane >> 4;
    const float* p = x0 + (size_t)row * ldk + k0 + 8 * h;
    float4 a0 = *(const float4*)p;
    float4 a1 = *(const float4*)(p + 4);
    float4 b0 = *(const float4*)(p + 16);
    float4 b1 = *(const float4*)(p + 20);
    Frag f;
    f.u[0] = pack2(a0.x, a0.y); f.u[1] = pack2(a0.z, a0.w);
    f.u[2] = pack2(a1.x, a1.y); f.u[3] = pack2(a1.z, a1.w);
    f.u[4] = pack2(b0.x, b0.y); f.u[5] = pack2(b0.z, b0.w);
    f.u[6] = pack2(b1.x, b1.y); f.u[7] = pack2(b1.z, b1.w);
    return f;
}

// store one C-row element to a bf16 row-major LDS tile, packing neighbor
// columns via lane-pair shuffle: even lanes store one b32 covering 2 columns.
__device__ __forceinline__ void store_row_pair(unsigned short* T, int ldk, int n0,
                                               float v, int r, int lane) {
    int lr = lane & 15, half = lane >> 4;
    float o = __shfl_xor(v, 1, 32);
    if ((lr & 1) == 0) {
        unsigned int pk = pack2(v, o);
        int M = r + 8 * half;
        *(unsigned int*)(T + M * ldk + n0 + lr) = pk;
    }
}

// ---------------- Kernel A: weight transpose/convert + accumulator zero ----------------
__global__ void moe_prep_kernel(const float* __restrict__ pe_w1,
                                const float* __restrict__ pe_w2,
                                const float* __restrict__ ew1,
                                const float* __restrict__ ew2,
                                unsigned short* __restrict__ pw1t,
                                unsigned short* __restrict__ pw2t,
                                unsigned short* __restrict__ ew1t,
                                unsigned short* __restrict__ ew2t,
                                float* __restrict__ zacc,
                                float* __restrict__ pfacc) {
    int i = blockIdx.x * blockDim.x + threadIdx.x;
    if (i < DD * INN) {                 // pw1t[n][k] = pe_w1[k][n]
        int n = i / INN, k = i % INN;
        pw1t[i] = f2bf(pe_w1[k * DD + n]);
    }
    if (i < DD * DD) {                  // pw2t[n][k] = pe_w2[k][n]
        int n = i / DD, k = i % DD;
        pw2t[i] = f2bf(pe_w2[k * DD + n]);
    }
    if (i < EE * HIDD * DD) {           // ew1t[e][h][d] = ew1[e][d][h]
        int e = i / (HIDD * DD), r = i % (HIDD * DD);
        int hh = r / DD, d = r % DD;
        ew1t[i] = f2bf(ew1[e * DD * HIDD + d * HIDD + hh]);
    }
    if (i < EE * DD * HIDD) {           // ew2t[e][d][h] = ew2[e][h][d]
        int e = i / (DD * HIDD), r = i % (DD * HIDD);
        int d = r / HIDD, hh = r % HIDD;
        ew2t[i] = f2bf(ew2[e * HIDD * DD + hh * DD + d]);
    }
    if (i < BATCHH * DD) zacc[i] = 0.0f;
    if (i < 8) pfacc[i] = 0.0f;
}

// ---------------- Kernel B: fused encoder + gate + routed experts ----------------
__global__ __launch_bounds__(128, 1)
void moe_main_kernel(const float* __restrict__ x,
                     const unsigned short* __restrict__ pw1t,   // pw2t contiguous after
                     const float* __restrict__ pe_b1,
                     const float* __restrict__ pe_b2,
                     const float* __restrict__ gate_w,
                     const float* __restrict__ gate_b,
                     const unsigned short* __restrict__ ew1t,
                     const float* __restrict__ eb1,
                     const unsigned short* __restrict__ ew2t,
                     const float* __restrict__ eb2,
                     float* __restrict__ zacc,
                     float* __restrict__ pfacc) {
    // LDS budget: 8KB h + 16KB eh + 32KB weights + 0.5KB gate = ~56.5KB (< 64KB)
    __shared__ __align__(16) unsigned short hbuf[4][16 * DD];     // h tile bf16 (per wave)
    __shared__ __align__(16) unsigned short ehbuf[4][16 * 128];   // expert hidden half-tile
    __shared__ __align__(16) unsigned short wlds[12288 + 4096];   // pe_w1^T | pe_w2^T bf16
    __shared__ float gatev[4][16];
    __shared__ int   top1v[4][16];

    const int lane = threadIdx.x & 31;
    const int wave = threadIdx.x >> 5;
    const int half = lane >> 4;
    const int lr   = lane & 15;

    unsigned short* hb  = hbuf[wave];
    unsigned short* ehb = ehbuf[wave];
    float* gv  = gatev[wave];
    int*   tv  = top1v[wave];
    const unsigned short* w1l = wlds;            // [64][192]
    const unsigned short* w2l = wlds + 12288;    // [64][64]

    // ---- stage encoder weights (32KB, contiguous in workspace) into LDS ----
    {
        uint4* gsrc = (uint4*)pw1t;          // pw1t..pw2t are contiguous (32768 B)
        uint4* ldst = (uint4*)wlds;
#ifdef ASYNC_WLOAD
        for (int i = threadIdx.x; i < 2048; i += 128) {
            as1_v4i gp = (as1_v4i)(as1_void)(gsrc + i);
            as3_v4i lp = (as3_v4i)(as3_void)(ldst + i);
            __builtin_amdgcn_global_load_async_to_lds_b128(gp, lp, 0, 0);
        }
        asm volatile("s_wait_asynccnt 0x0" ::: "memory");
#else
        for (int i = threadIdx.x; i < 2048; i += 128) ldst[i] = gsrc[i];
#endif
        __syncthreads();
    }

    // warm the (global) expert-weight path into WGP caches
    __builtin_prefetch(ew1t + (size_t)lane * 1024, 0, 0);
    __builtin_prefetch(ew2t + (size_t)lane * 1024, 0, 0);

    // per-wave aux-loss accumulators (avoid hot-address atomics per tile)
    float paccl[4] = {0.f, 0.f, 0.f, 0.f};
    float faccl[4] = {0.f, 0.f, 0.f, 0.f};

    for (int tile = blockIdx.x * 4 + wave; tile < NTILEE; tile += gridDim.x * 4) {
        const float* xt = x + (size_t)tile * 16 * INN;
        const int batch = tile >> 6;   // tile*16 / 1024

        // ---- GEMM1: t1 = gelu(x @ pe_w1 + pe_b1), 16x192 @ 192x64 ----
        Frag ax[6];
        #pragma unroll
        for (int kk = 0; kk < 6; ++kk) ax[kk] = loadA_x(xt, INN, kk * 32, lane);

        #pragma unroll
        for (int nt = 0; nt < 4; ++nt) {
            v8f acc = {0.f, 0.f, 0.f, 0.f, 0.f, 0.f, 0.f, 0.f};
            #pragma unroll
            for (int kk = 0; kk < 6; ++kk)
                acc = wmma_bf16(ax[kk], loadB(w1l, INN, nt * 16, kk * 32, lane), acc);
            float bias = pe_b1[nt * 16 + lr];
            #pragma unroll
            for (int r = 0; r < 8; ++r)
                store_row_pair(hb, DD, nt * 16, gelu_exact(acc[r] + bias), r, lane);
        }
        lds_wave_fence();

        // ---- GEMM2: h = t1 @ pe_w2 + pe_b2, 16x64 @ 64x64 ----
        v8f hacc[4];
        {
            Frag a0 = loadA_lds(hb, DD, 0, lane);
            Frag a1 = loadA_lds(hb, DD, 32, lane);
            #pragma unroll
            for (int nt = 0; nt < 4; ++nt) {
                v8f acc = {0.f, 0.f, 0.f, 0.f, 0.f, 0.f, 0.f, 0.f};
                acc = wmma_bf16(a0, loadB(w2l, DD, nt * 16, 0, lane), acc);
                acc = wmma_bf16(a1, loadB(w2l, DD, nt * 16, 32, lane), acc);
                float bias = pe_b2[nt * 16 + lr];
                #pragma unroll
                for (int r = 0; r < 8; ++r) acc[r] += bias;
                hacc[nt] = acc;
            }
        }
        #pragma unroll
        for (int nt = 0; nt < 4; ++nt)
            #pragma unroll
            for (int r = 0; r < 8; ++r)
                store_row_pair(hb, DD, nt * 16, hacc[nt][r], r, lane);
        lds_wave_fence();

        // ---- gate: softmax + top-1 (lanes 0-15, one row each, f32 VALU) ----
        int myexp = -1;
        if (lane < 16) {
            const int row = lane;
            const unsigned int* hrow = (const unsigned int*)(hb + row * DD);
            float lg[4] = {gate_b[0], gate_b[1], gate_b[2], gate_b[3]};
            #pragma unroll 8
            for (int k2 = 0; k2 < DD / 2; ++k2) {
                unsigned int w = hrow[k2];
                float h0 = __builtin_bit_cast(float, w << 16);
                float h1 = __builtin_bit_cast(float, w & 0xffff0000u);
                const float* g0 = gate_w + (2 * k2) * EE;
                #pragma unroll
                for (int e = 0; e < 4; ++e) lg[e] += h0 * g0[e] + h1 * g0[EE + e];
            }
            float m = fmaxf(fmaxf(lg[0], lg[1]), fmaxf(lg[2], lg[3]));
            float ex[4], s = 0.f;
            #pragma unroll
            for (int e = 0; e < 4; ++e) { ex[e] = expf(lg[e] - m); s += ex[e]; }
            float inv = 1.0f / s;
            int t1i = 0; float best = lg[0];
            #pragma unroll
            for (int e = 1; e < 4; ++e)
                if (lg[e] > best) { best = lg[e]; t1i = e; }   // first-max tie-break
            gv[row] = ex[t1i] * inv;
            tv[row] = t1i;
            myexp = t1i;
            #pragma unroll
            for (int e = 0; e < 4; ++e) paccl[e] += ex[e] * inv;
            faccl[t1i] += 1.0f;
        }
        lds_wave_fence();

        int tvr[8]; float gvr[8];
        #pragma unroll
        for (int r = 0; r < 8; ++r) {
            int M = r + 8 * half;
            tvr[r] = tv[M];
            gvr[r] = gv[M];
        }

        // hb is constant through the expert loop: hoist its A-fragments
        Frag ha0 = loadA_lds(hb, DD, 0, lane);
        Frag ha1 = loadA_lds(hb, DD, 32, lane);

        // ---- routed experts (skip experts no token in this tile selected) ----
        float macc[4][8];
        #pragma unroll
        for (int nt = 0; nt < 4; ++nt)
            #pragma unroll
            for (int r = 0; r < 8; ++r) macc[nt][r] = 0.f;

        for (int e = 0; e < EE; ++e) {
            if (__builtin_amdgcn_ballot_w32(myexp == e) == 0u) continue;  // uniform skip
            const unsigned short* W1 = ew1t + (size_t)e * HIDD * DD;
            const unsigned short* W2 = ew2t + (size_t)e * DD * HIDD;
            const float* B1 = eb1 + e * HIDD;
            const float* B2 = eb2 + e * DD;

            v8f acc4[4];   // eo accumulators, carried across the two K-halves
            #pragma unroll
            for (int nt = 0; nt < 4; ++nt)
                acc4[nt] = (v8f){0.f, 0.f, 0.f, 0.f, 0.f, 0.f, 0.f, 0.f};

            #pragma unroll
            for (int hh = 0; hh < 2; ++hh) {
                // eh(:, hh*128 .. hh*128+127) = gelu(h @ ew1[e] + eb1[e])
                #pragma unroll
                for (int nt = 0; nt < 8; ++nt) {
                    int gcol = (hh * 8 + nt) * 16;
                    v8f acc = {0.f, 0.f, 0.f, 0.f, 0.f, 0.f, 0.f, 0.f};
                    acc = wmma_bf16(ha0, loadB(W1, DD, gcol, 0, lane), acc);
                    acc = wmma_bf16(ha1, loadB(W1, DD, gcol, 32, lane), acc);
                    float bias = B1[gcol + lr];
                    #pragma unroll
                    for (int r = 0; r < 8; ++r)
                        store_row_pair(ehb, 128, nt * 16,
                                       gelu_exact(acc[r] + bias), r, lane);
                }
                lds_wave_fence();

                // eo += eh_half @ ew2[e][:, hh*128 : hh*128+128]
                Frag ae[4];
                #pragma unroll
                for (int kk = 0; kk < 4; ++kk)
                    ae[kk] = loadA_lds(ehb, 128, kk * 32, lane);
                #pragma unroll
                for (int nt = 0; nt < 4; ++nt)
                    #pragma unroll
                    for (int kk = 0; kk < 4; ++kk)
                        acc4[nt] = wmma_bf16(ae[kk],
                                             loadB(W2, HIDD, nt * 16,
                                                   hh * 128 + kk * 32, lane),
                                             acc4[nt]);
                lds_wave_fence();   // before next half overwrites ehb
            }

            #pragma unroll
            for (int nt = 0; nt < 4; ++nt) {
                float bias = B2[nt * 16 + lr];
                #pragma unroll
                for (int r = 0; r < 8; ++r) {
                    if (tvr[r] == e) macc[nt][r] += gvr[r] * (acc4[nt][r] + bias);
                }
            }
        }

        // ---- accumulate sequence-sum of moe_out into z[batch] ----
        #pragma unroll
        for (int nt = 0; nt < 4; ++nt) {
            float s = 0.f;
            #pragma unroll
            for (int r = 0; r < 8; ++r) s += macc[nt][r];
            atomicAdd(&zacc[batch * DD + nt * 16 + lr], s);
        }
    }

    // ---- aux-loss sums: one shuffle-tree reduction + <=8 atomics per wave ----
    #pragma unroll
    for (int i = 0; i < 8; ++i) {
        float v = (i < 4) ? paccl[i] : faccl[i - 4];
        v += __shfl_xor(v, 16, 32);
        v += __shfl_xor(v, 8, 32);
        v += __shfl_xor(v, 4, 32);
        v += __shfl_xor(v, 2, 32);
        v += __shfl_xor(v, 1, 32);
        if (lane == 0) atomicAdd(&pfacc[i], v);
    }
}

// ---------------- Kernel C: mean-pool finalize + LayerNorm + classifier + aux ----------------
__global__ __launch_bounds__(128)
void moe_head_kernel(const float* __restrict__ zacc,
                     const float* __restrict__ pfacc,
                     const float* __restrict__ ln_g, const float* __restrict__ ln_b,
                     const float* __restrict__ cl_w1, const float* __restrict__ cl_b1,
                     const float* __restrict__ cl_w2, const float* __restrict__ cl_b2,
                     float* __restrict__ out) {
    int row = threadIdx.x;
    if (row < BATCHH) {
        float z[DD];
        float mu = 0.f;
        #pragma unroll
        for (int d = 0; d < DD; ++d) {
            z[d] = zacc[row * DD + d] * (1.0f / SEQQ);
            mu += z[d];
        }
        mu *= (1.0f / DD);
        float var = 0.f;
        #pragma unroll
        for (int d = 0; d < DD; ++d) { float t = z[d] - mu; var += t * t; }
        var *= (1.0f / DD);
        float rs = rsqrtf(var + 1e-5f);
        float zn[DD];
        #pragma unroll
        for (int d = 0; d < DD; ++d) zn[d] = (z[d] - mu) * rs * ln_g[d] + ln_b[d];
        float t1[DD];
        for (int j = 0; j < DD; ++j) {
            float a = cl_b1[j];
            #pragma unroll
            for (int d = 0; d < DD; ++d) a += zn[d] * cl_w1[d * DD + j];
            t1[j] = gelu_exact(a);
        }
        for (int j = 0; j < NCLSS; ++j) {
            float a = cl_b2[j];
            #pragma unroll
            for (int d = 0; d < DD; ++d) a += t1[d] * cl_w2[d * NCLSS + j];
            out[row * NCLSS + j] = a;
        }
    }
    if (threadIdx.x == 0 && blockIdx.x == 0) {
        float aux = 0.f;
        #pragma unroll
        for (int e = 0; e < EE; ++e)
            aux += (pfacc[4 + e] * (1.0f / NTOKK)) * (pfacc[e] * (1.0f / NTOKK));
        out[BATCHH * NCLSS] = (float)EE * aux;
    }
}

extern "C" void kernel_launch(void* const* d_in, const int* in_sizes, int n_in,
                              void* d_out, int out_size, void* d_ws, size_t ws_size,
                              hipStream_t stream) {
    const float* x      = (const float*)d_in[0];
    const float* pe_w1  = (const float*)d_in[1];
    const float* pe_b1  = (const float*)d_in[2];
    const float* pe_w2  = (const float*)d_in[3];
    const float* pe_b2  = (const float*)d_in[4];
    const float* gate_w = (const float*)d_in[5];
    const float* gate_b = (const float*)d_in[6];
    const float* ew1    = (const float*)d_in[7];
    const float* eb1    = (const float*)d_in[8];
    const float* ew2    = (const float*)d_in[9];
    const float* eb2    = (const float*)d_in[10];
    const float* ln_g   = (const float*)d_in[11];
    const float* ln_b   = (const float*)d_in[12];
    const float* cl_w1  = (const float*)d_in[13];
    const float* cl_b1  = (const float*)d_in[14];
    const float* cl_w2  = (const float*)d_in[15];
    const float* cl_b2  = (const float*)d_in[16];

    char* ws = (char*)d_ws;
    float* zacc  = (float*)(ws + OFF_Z);
    float* pfacc = (float*)(ws + OFF_ACC);
    unsigned short* pw1t = (unsigned short*)(ws + OFF_PW1T);
    unsigned short* pw2t = (unsigned short*)(ws + OFF_PW2T);
    unsigned short* ew1t = (unsigned short*)(ws + OFF_EW1T);
    unsigned short* ew2t = (unsigned short*)(ws + OFF_EW2T);

    moe_prep_kernel<<<256, 256, 0, stream>>>(pe_w1, pe_w2, ew1, ew2,
                                             pw1t, pw2t, ew1t, ew2t, zacc, pfacc);

    moe_main_kernel<<<512, 128, 0, stream>>>(x, pw1t, pe_b1, pe_b2,
                                             gate_w, gate_b, ew1t, eb1, ew2t, eb2,
                                             zacc, pfacc);

    moe_head_kernel<<<1, 128, 0, stream>>>(zacc, pfacc, ln_g, ln_b,
                                           cl_w1, cl_b1, cl_w2, cl_b2,
                                           (float*)d_out);
}
